// LSTMPredictor_22368189677727
// MI455X (gfx1250) — matrix operand order repaired and checked by reading, not compiled, not run on hardware
//
#include <hip/hip_runtime.h>

// Problem constants (from the reference): B=4096, T=512, IN=1, H=32, OUT=1
#define NB 4096
#define NT 512
#define NH 32

typedef __attribute__((ext_vector_type(16))) __bf16 v16bf;
typedef __attribute__((ext_vector_type(8)))  __bf16 v8bf;
typedef __attribute__((ext_vector_type(8)))  float  v8f;

static __device__ __forceinline__ float ftanh(float x) {
#if __has_builtin(__builtin_amdgcn_tanhf)
  return __builtin_amdgcn_tanhf(x);          // gfx1250 v_tanh_f32
#else
  float e = __builtin_amdgcn_exp2f(-2.885390082f * x); // exp(-2x)
  return (1.0f - e) * __builtin_amdgcn_rcpf(1.0f + e);
#endif
}
// Sigmoid of (2*a) when the pre-activation was already scaled by 0.5:
// sigmoid(x) = 0.5 + 0.5*tanh(x/2); here acc == x/2 already.
static __device__ __forceinline__ float fsig_pre(float acc_half) {
  return fmaf(ftanh(acc_half), 0.5f, 0.5f);
}

// ---------------------------------------------------------------------------
// Kernel 1: transpose x[B,T] -> xT[T,B] so the recurrence reads x coalesced.
// ---------------------------------------------------------------------------
__global__ void xpose(const float* __restrict__ x, float* __restrict__ xT) {
  __shared__ float tile[32][33];
  int b0 = blockIdx.x * 32, t0 = blockIdx.y * 32;
  int lx = threadIdx.x, ly = threadIdx.y;  // block 32x8
#pragma unroll
  for (int k = 0; k < 4; ++k) {
    int i = ly + 8 * k;
    tile[i][lx] = x[(b0 + i) * NT + t0 + lx];
  }
  __syncthreads();
#pragma unroll
  for (int k = 0; k < 4; ++k) {
    int i = ly + 8 * k;
    xT[(t0 + i) * NB + b0 + lx] = tile[lx][i];
  }
}

// ---------------------------------------------------------------------------
// Kernel 2: recurrence. One wave owns 16 batch rows; per step 8 WMMAs
// (v_wmma_f32_16x16x32_bf16, full K=32 per instruction) compute
// gates[16,128], with C = x*W_ih + (b_ih+b_hh) preloaded in the accumulator.
// i/f/o tiles (j = 0,1,2,3,6,7) have W_hh/W_ih/bias pre-scaled by 0.5 so
// sigmoid is a single tanh + fma; g tiles (j = 4,5) are unscaled.
// h is round-tripped through a wave-private 16x32 bf16 LDS tile
// (row stride 80B -> conflict-free b128 reloads) for the D->A relayout.
// ---------------------------------------------------------------------------
__global__ void __launch_bounds__(64) lstm_rec(
    const float* __restrict__ xT,
    const float* __restrict__ W_ih, const float* __restrict__ W_hh,
    const float* __restrict__ b_ih, const float* __restrict__ b_hh,
    const float* __restrict__ fc_w, const float* __restrict__ fc_b,
    float* __restrict__ out)
{
  __shared__ __align__(16) __bf16 hsm[2 * 16 * 40]; // 2 waves x (16 rows x 40)
  const int lane    = threadIdx.x & 31;
  const int wib     = threadIdx.x >> 5;
  const int rowBase = (blockIdx.x * 2 + wib) * 16;
  const int col = lane & 15;      // N column within 16-wide tile / A row M
  const int hi  = lane >> 4;      // lane half
  const int m0  = hi * 8;         // C/D row offset for this lane half
  const int k0  = hi * 8;         // A-layout K base for this lane half
  const int k0b = hi * 16;        // contiguous K base (k = 16h+s) used here for both A and B
  __bf16* wsm = hsm + wib * (16 * 40);

  // --- one-time: per-lane gate constants + B tiles (W_hh.T) in registers ---
  float wih[8], bia[8];
  v16bf Bm[8];
#pragma unroll
  for (int j = 0; j < 8; ++j) {
    const float sc = (j == 4 || j == 5) ? 1.0f : 0.5f;  // g-gate unscaled
    int g = 16 * j + col;                 // gate column 16j+n
    wih[j] = W_ih[g] * sc;
    bia[j] = (b_ih[g] + b_hh[g]) * sc;
    union { v16bf v; __bf16 s[16]; } bu;
#pragma unroll
    for (int e = 0; e < 16; ++e)          // B[k,n] = W_hh[g,k], k = k0b+e
      bu.s[e] = (__bf16)(W_hh[g * NH + k0b + e] * sc);
    Bm[j] = bu.v;
  }

  // h0 = 0 -> A starts as zeros
  union { v16bf v; unsigned int u[8]; } az;
#pragma unroll
  for (int e = 0; e < 8; ++e) az.u[e] = 0u;
  v16bf A = az.v;

  float cst0[8], cst1[8], hv0[8], hv1[8];
#pragma unroll
  for (int r = 0; r < 8; ++r) { cst0[r] = 0.f; cst1[r] = 0.f; hv0[r] = 0.f; hv1[r] = 0.f; }

  // double-buffered x fetch: 8 floats (rows m0..m0+7) per step
  const float* xrow = xT + rowBase + m0;
  float4 xa = *(const float4*)(xrow);
  float4 xb = *(const float4*)(xrow + 4);

#pragma unroll 1
  for (int t = 0; t < NT; ++t) {
    int tn = (t + 1 < NT) ? (t + 1) : t;
    const float* xn = xT + (long)tn * NB + rowBase + m0;
    float4 na = *(const float4*)(xn);
    float4 nb = *(const float4*)(xn + 4);

    float xv[8] = {xa.x, xa.y, xa.z, xa.w, xb.x, xb.y, xb.z, xb.w};

    // gates = h @ W_hh.T + (x*W_ih + bias): 8 WMMAs, K=32 in one shot each
    v8f acc[8];
#pragma unroll
    for (int j = 0; j < 8; ++j) {
      v8f C;
#pragma unroll
      for (int r = 0; r < 8; ++r) C[r] = fmaf(xv[r], wih[j], bia[j]);
      acc[j] = __builtin_amdgcn_wmma_f32_16x16x32_bf16(
                   false, A, false, Bm[j], (short)0, C, false, false);
    }

    // tiles: 0,1 = i | 2,3 = f | 4,5 = g | 6,7 = o  (cols 0-15 / 16-31)
#pragma unroll
    for (int r = 0; r < 8; ++r) {
      {
        float iv = fsig_pre(acc[0][r]), fv = fsig_pre(acc[2][r]);
        float gv = ftanh(acc[4][r]),    ov = fsig_pre(acc[6][r]);
        float c = fmaf(fv, cst0[r], iv * gv);
        cst0[r] = c;
        float h = ov * ftanh(c);
        hv0[r] = h;
        wsm[(m0 + r) * 40 + col] = (__bf16)h;      // native f32->bf16 cvt
      }
      {
        float iv = fsig_pre(acc[1][r]), fv = fsig_pre(acc[3][r]);
        float gv = ftanh(acc[5][r]),    ov = fsig_pre(acc[7][r]);
        float c = fmaf(fv, cst1[r], iv * gv);
        cst1[r] = c;
        float h = ov * ftanh(c);
        hv1[r] = h;
        wsm[(m0 + r) * 40 + col + 16] = (__bf16)h;
      }
    }

    // rebuild A (16x32 bf16, A-layout) from the row-major LDS h tile:
    // lane reads row `col`, K chunks [k0b..k0b+7] and [k0b+8..k0b+15]
    v8bf c0 = *reinterpret_cast<const v8bf*>(wsm + col * 40 + k0b);
    v8bf c1 = *reinterpret_cast<const v8bf*>(wsm + col * 40 + k0b + 8);
    A = __builtin_shufflevector(c0, c1, 0,1,2,3,4,5,6,7,8,9,10,11,12,13,14,15);

    xa = na; xb = nb;
  }

  // --- final FC: out[b] = h_last[b,:] . fc_w + fc_b (half-wave reduction) ---
  float fw0 = fc_w[col], fw1 = fc_w[col + 16], fb = fc_b[0];
#pragma unroll
  for (int r = 0; r < 8; ++r) {
    float part = fmaf(hv0[r], fw0, hv1[r] * fw1);
#pragma unroll
    for (int off = 1; off < 16; off <<= 1)
      part += __shfl_xor(part, off, 32);
    if (col == 0)
      out[rowBase + m0 + r] = part + fb;
  }
}

// ---------------------------------------------------------------------------
extern "C" void kernel_launch(void* const* d_in, const int* in_sizes, int n_in,
                              void* d_out, int out_size, void* d_ws, size_t ws_size,
                              hipStream_t stream) {
  const float* x    = (const float*)d_in[0];
  const float* W_ih = (const float*)d_in[1];
  const float* W_hh = (const float*)d_in[2];
  const float* b_ih = (const float*)d_in[3];
  const float* b_hh = (const float*)d_in[4];
  const float* fc_w = (const float*)d_in[5];
  const float* fc_b = (const float*)d_in[6];
  float* outp = (float*)d_out;
  float* xT   = (float*)d_ws;               // NT*NB floats = 8 MB scratch

  dim3 gT(NB / 32, NT / 32), bT(32, 8);
  xpose<<<gT, bT, 0, stream>>>(x, xT);

  // 4096 rows / (2 waves * 16 rows) = 128 blocks of 64 threads
  lstm_rec<<<NB / 32, 64, 0, stream>>>(xT, W_ih, W_hh, b_ih, b_hh,
                                       fc_w, fc_b, outp);
}
